// BetaEncoder_16123307229255
// MI455X (gfx1250) — 16-bit and 8-bit matrix operand order repaired and checked by reading, compiled, not run on hardware
//
#include <hip/hip_runtime.h>

#define B_  16
#define T_  4096
#define P_  256
#define W_  512
#define G3  (3 * W_)

typedef __attribute__((ext_vector_type(16))) __bf16 v16bf;
typedef __attribute__((ext_vector_type(8)))  __bf16 v8bf;
typedef __attribute__((ext_vector_type(8)))  float  v8f;
typedef __attribute__((ext_vector_type(16))) int    v16i;
typedef __attribute__((ext_vector_type(4)))  int    v4i;

union AFrag { v16bf v; v8bf h[2]; };
union AF8   { v16i  v; unsigned long long d[8]; };
union BF8   { v16i  v; v4i q[4]; };

__device__ __forceinline__ unsigned short f2bf(float f) {
    unsigned int u = __float_as_uint(f);
    u += 0x7FFFu + ((u >> 16) & 1u);
    return (unsigned short)(u >> 16);
}

__device__ __forceinline__ unsigned char f2fp8(float x) {
    unsigned int u   = __float_as_uint(x);
    unsigned int sgn = (u >> 24) & 0x80u;
    int          e   = (int)((u >> 23) & 0xFFu) - 127;
    unsigned int man = u & 0x7FFFFFu;
    if (e > 8) return (unsigned char)(sgn | 0x7Eu);
    if (e >= -6) {
        unsigned int m    = man >> 20;
        unsigned int rest = man & 0xFFFFFu;
        m += (rest > 0x80000u) || (rest == 0x80000u && (m & 1u));
        unsigned int code = ((unsigned int)(e + 7) << 3) + m;
        if (code > 0x7Eu) code = 0x7Eu;
        return (unsigned char)(sgn | code);
    }
    if (e < -10) return (unsigned char)sgn;
    unsigned int full = 0x800000u | man;
    int          sh   = 14 - e;
    unsigned int m    = full >> sh;
    unsigned int rem  = full & ((1u << sh) - 1u);
    unsigned int half = 1u << (sh - 1);
    m += (rem > half) || (rem == half && (m & 1u));
    return (unsigned char)(sgn | m);
}

__global__ void cvt_f32_bf16(const float* __restrict__ s,
                             unsigned short* __restrict__ d, int n) {
    int i = blockIdx.x * blockDim.x + threadIdx.x;
    if (i < n) d[i] = f2bf(s[i]);
}

__global__ void cvt_f32_fp8(const float* __restrict__ s,
                            unsigned char* __restrict__ d, int n) {
    int i = blockIdx.x * blockDim.x + threadIdx.x;
    if (i < n) d[i] = f2fp8(s[i]);
}

__global__ void cvt_a(const float* __restrict__ a,
                      unsigned short* __restrict__ d) {
    long i = (long)blockIdx.x * blockDim.x + threadIdx.x;
    int  p  = (int)(i & (P_ - 1));
    long tb = i >> 8;
    int  b  = (int)(tb & 15);
    long t  = tb >> 4;
    d[i] = f2bf(a[((long)b * T_ + t) * P_ + p]);
}

__global__ void __launch_bounds__(256)
ig_gemm(const unsigned short* __restrict__ A,
        const unsigned short* __restrict__ Wih,
        const float* __restrict__ bias,
        float* __restrict__ IG) {
    int lane = threadIdx.x & 31;
    int wave = threadIdx.x >> 5;
    int gid  = blockIdx.x * 8 + wave;
    int mt   = gid / 12;
    int ng   = gid - mt * 12;
    int col    = lane & 15;
    int rowoff = (lane >> 4) << 3;
    int ksel   = (lane >= 16) ? 8  : 0;
    int klane  = (lane >= 16) ? 16 : 0;
    const unsigned short* arow  = A   + ((long)mt * 16 + col) * P_;
    const unsigned short* bbase = Wih + ((long)(ng * 8) * 16 + col) * P_;
    v8f acc[8] = {};
#pragma unroll
    for (int k0 = 0; k0 < P_; k0 += 32) {
        AFrag af;
        af.h[0] = *(const v8bf*)(arow + k0 + klane);
        af.h[1] = *(const v8bf*)(arow + k0 + klane + 8);
#pragma unroll
        for (int i = 0; i < 8; ++i) {
            v16bf bv = *(const v16bf*)(bbase + (long)i * 16 * P_ + k0 + klane);
            acc[i] = __builtin_amdgcn_wmma_f32_16x16x32_bf16(
                         false, af.v, false, bv, (short)0, acc[i], false, false);
        }
    }
#pragma unroll
    for (int i = 0; i < 8; ++i) {
        int nc = (ng * 8 + i) * 16 + col;
        float bb = bias[nc];
        float* o = IG + ((long)mt * 16 + rowoff) * G3 + nc;
#pragma unroll
        for (int r = 0; r < 8; ++r) o[(long)r * G3] = acc[i][r] + bb;
    }
}

__global__ void __launch_bounds__(1024)
gru_scan(const float* __restrict__ IG,
         const unsigned char* __restrict__ Whh,
         const float* __restrict__ h0,
         const float* __restrict__ bn,
         unsigned short* __restrict__ HS) {
    __shared__ unsigned char h8[2][B_][W_];

    int lane   = threadIdx.x & 31;
    int wave   = threadIdx.x >> 5;
    int col    = lane & 15;
    int rowoff = (lane >> 4) << 3;
    int ks     = (lane >= 16) ? 8  : 0;
    int kb     = (lane >= 16) ? 16 : 0;
    int j      = wave * 16 + col;

    const unsigned char* brR = Whh + ((long)(wave     ) * 16 + col) * W_;
    const unsigned char* brZ = Whh + ((long)(wave + 32) * 16 + col) * W_;
    const unsigned char* brN = Whh + ((long)(wave + 64) * 16 + col) * W_;

    float bnj = bn[j];
    float h0j = h0[j];
    float h[8];
#pragma unroll
    for (int r = 0; r < 8; ++r) h[r] = h0j;
    unsigned char h0q = f2fp8(h0j);
#pragma unroll
    for (int r = 0; r < 8; ++r) h8[0][rowoff + r][j] = h0q;
    __syncthreads();

    int buf = 0;
    for (int t = T_ - 1; t >= 0; --t) {
        const unsigned char* hrow = &h8[buf][0][0] + (long)col * W_;
        v8f aR = {}, aZ = {}, aN = {};
#pragma unroll
        for (int k0 = 0; k0 < W_; k0 += 128) {
            AF8 af;
#pragma unroll
            for (int c = 0; c < 8; ++c)
                af.d[c] = *(const unsigned long long*)
                          (hrow + k0 + ((c >> 1) * 32) + ((c & 1) * 8) + kb);
            BF8 bR, bZ, bN;
#pragma unroll
            for (int q = 0; q < 4; ++q) {
                bR.q[q] = *(const v4i*)(brR + k0 + q * 32 + kb);
                bZ.q[q] = *(const v4i*)(brZ + k0 + q * 32 + kb);
                bN.q[q] = *(const v4i*)(brN + k0 + q * 32 + kb);
            }
            aR = __builtin_amdgcn_wmma_f32_16x16x128_fp8_fp8(
                     af.v, bR.v, (short)0, aR, false, false);
            aZ = __builtin_amdgcn_wmma_f32_16x16x128_fp8_fp8(
                     af.v, bZ.v, (short)0, aZ, false, false);
            aN = __builtin_amdgcn_wmma_f32_16x16x128_fp8_fp8(
                     af.v, bN.v, (short)0, aN, false, false);
        }
        const float* ig = IG + ((long)t * 16 + rowoff) * G3;
        if (t)
            __builtin_prefetch(IG + ((long)(t - 1) * 16) * G3 + wave * 16, 0, 0);

        unsigned char*  hw = &h8[buf ^ 1][0][0];
        unsigned short* hs = HS + ((long)t * 16) * W_;
#pragma unroll
        for (int r = 0; r < 8; ++r) {
            const float* igrow = ig + (long)r * G3;
            float xr = igrow[j];
            float xz = igrow[W_ + j];
            float xn = igrow[2 * W_ + j];
            float rg = 1.0f / (1.0f + __expf(-(xr + aR[r])));
            float zg = 1.0f / (1.0f + __expf(-(xz + aZ[r])));
            float ng = tanhf(xn + rg * (aN[r] + bnj));
            float hn = ng + zg * (h[r] - ng);
            h[r] = hn;
            hw[(long)(rowoff + r) * W_ + j] = f2fp8(hn);
            hs[(long)(rowoff + r) * W_ + j] = f2bf(hn);
        }
        __syncthreads();
        buf ^= 1;
    }
}

__global__ void __launch_bounds__(256)
out_gemm(const unsigned short* __restrict__ HS,
         const unsigned short* __restrict__ Wout,
         const float* __restrict__ bout,
         float* __restrict__ out) {
    int lane = threadIdx.x & 31;
    int wave = threadIdx.x >> 5;
    int gid  = blockIdx.x * 8 + wave;
    int mt   = gid >> 2;
    int ng   = gid & 3;
    int col    = lane & 15;
    int rowoff = (lane >> 4) << 3;
    int ksel   = (lane >= 16) ? 8  : 0;
    int klane  = (lane >= 16) ? 16 : 0;
    const unsigned short* arow  = HS   + ((long)mt * 16 + col) * W_;
    const unsigned short* bbase = Wout + ((long)(ng * 4) * 16 + col) * W_;
    v8f acc[4] = {};
#pragma unroll 4
    for (int k0 = 0; k0 < W_; k0 += 32) {
        AFrag af;
        af.h[0] = *(const v8bf*)(arow + k0 + klane);
        af.h[1] = *(const v8bf*)(arow + k0 + klane + 8);
#pragma unroll
        for (int i = 0; i < 4; ++i) {
            v16bf bv = *(const v16bf*)(bbase + (long)i * 16 * W_ + k0 + klane);
            acc[i] = __builtin_amdgcn_wmma_f32_16x16x32_bf16(
                         false, af.v, false, bv, (short)0, acc[i], false, false);
        }
    }
#pragma unroll
    for (int i = 0; i < 4; ++i) {
        int nc = (ng * 4 + i) * 16 + col;
        float bb = bout[nc];
#pragma unroll
        for (int r = 0; r < 8; ++r) {
            int b = rowoff + r;
            out[(long)b * T_ * P_ + (long)mt * P_ + nc] = acc[i][r] + bb;
        }
    }
}

extern "C" void kernel_launch(void* const* d_in, const int* in_sizes, int n_in,
                              void* d_out, int out_size, void* d_ws, size_t ws_size,
                              hipStream_t stream) {
    const float* a     = (const float*)d_in[0];
    const float* h0    = (const float*)d_in[1];
    const float* w_ih  = (const float*)d_in[2];
    const float* w_hh  = (const float*)d_in[3];
    const float* bias  = (const float*)d_in[4];
    const float* bn    = (const float*)d_in[5];
    const float* w_out = (const float*)d_in[6];
    const float* b_out = (const float*)d_in[7];
    float* out = (float*)d_out;

    char* ws = (char*)d_ws;
    auto carve = [&](size_t bytes) -> char* {
        char* p = ws;
        ws += (bytes + 255) & ~(size_t)255;
        return p;
    };
    unsigned short* a_bf    = (unsigned short*)carve((size_t)B_ * T_ * P_ * 2);
    unsigned short* wih_bf  = (unsigned short*)carve((size_t)G3 * P_ * 2);
    unsigned char*  whh_f8  = (unsigned char*) carve((size_t)G3 * W_);
    unsigned short* wout_bf = (unsigned short*)carve((size_t)P_ * W_ * 2);
    float*          IG      = (float*)carve((size_t)B_ * T_ * G3 * 4);
    unsigned short* HS      = (unsigned short*)carve((size_t)B_ * T_ * W_ * 2);

    cvt_f32_bf16<<<(G3 * P_ + 255) / 256, 256, 0, stream>>>(w_ih,  wih_bf,  G3 * P_);
    cvt_f32_fp8 <<<(G3 * W_ + 255) / 256, 256, 0, stream>>>(w_hh,  whh_f8,  G3 * W_);
    cvt_f32_bf16<<<(P_ * W_ + 255) / 256, 256, 0, stream>>>(w_out, wout_bf, P_ * W_);
    cvt_a<<<((long)B_ * T_ * P_) / 256, 256, 0, stream>>>(a, a_bf);

    ig_gemm <<<(4096 * 12) / 8, 256, 0, stream>>>(a_bf, wih_bf, bias, IG);
    gru_scan<<<1, 1024, 0, stream>>>(IG, whh_f8, h0, bn, HS);
    out_gemm<<<(4096 * 4) / 8, 256, 0, stream>>>(HS, wout_bf, b_out, out);
}
